// VQVAE_42700564857164
// MI455X (gfx1250) — compile-verified, matrix operand order checked by reading, not run on hardware
//
#include <hip/hip_runtime.h>

#define BN_TOT  16384      // B*N = 16*1024 rows of x
#define CDIM    1024       // feature dim
#define KCB     8192       // codebook entries
#define OUT2OFF ((size_t)BN_TOT * CDIM)

typedef __attribute__((ext_vector_type(16))) __bf16 v16bf;
typedef __attribute__((ext_vector_type(8)))  float  v8f;

// =====================================================================
// Kernel A (fast path): one pass over codebook -> bf16 hi/lo split + cb_sq
//   grid = KCB blocks, 256 threads; each block handles one row (1024 f32)
// =====================================================================
__global__ void vq_split_cb_kernel(const float* __restrict__ cb,
                                   __bf16* __restrict__ cbHi,
                                   __bf16* __restrict__ cbLo,
                                   float* __restrict__ cbsq) {
    __shared__ float red[256];
    const int row = blockIdx.x;
    const int tid = threadIdx.x;

    float4 v = reinterpret_cast<const float4*>(cb + (size_t)row * CDIM)[tid];
    float f[4] = {v.x, v.y, v.z, v.w};
    const size_t off = (size_t)row * CDIM + tid * 4;
    float ss = 0.f;
#pragma unroll
    for (int e = 0; e < 4; ++e) {
        __bf16 h = (__bf16)f[e];
        cbHi[off + e] = h;
        cbLo[off + e] = (__bf16)(f[e] - (float)h);
        ss += f[e] * f[e];
    }
    red[tid] = ss;
    __syncthreads();
#pragma unroll
    for (int s = 128; s > 0; s >>= 1) {
        if (tid < s) red[tid] += red[tid + s];
        __syncthreads();
    }
    if (tid == 0) cbsq[row] = red[0];
}

// ---------------- cb_sq only (fallback path) ----------------
__global__ void vq_cbsq_kernel(const float* __restrict__ cb, float* __restrict__ cbsq) {
    const int wave = threadIdx.x >> 5;
    const int lane = threadIdx.x & 31;
    const int row  = blockIdx.x * 8 + wave;
    const float4* p4 = reinterpret_cast<const float4*>(cb + (size_t)row * CDIM);
    float s = 0.f;
#pragma unroll
    for (int i = lane; i < CDIM / 4; i += 32) {
        float4 v = p4[i];
        s += v.x * v.x + v.y * v.y + v.z * v.z + v.w * v.w;
    }
#pragma unroll
    for (int off = 16; off > 0; off >>= 1) s += __shfl_xor(s, off, 32);
    if (lane == 0) cbsq[row] = s;
}

// =====================================================================
// Shared device helpers for the main kernels
// =====================================================================
__device__ __forceinline__ void stage_x_tile(const float* __restrict__ x, int mBase,
                                             __bf16* sXhi, __bf16* sXlo, int tid) {
    for (int i = tid; i < 16 * (CDIM / 4); i += 256) {
        const int row = i >> 8;
        const int c4  = i & 255;
        float4 v = reinterpret_cast<const float4*>(x + (size_t)(mBase + row) * CDIM)[c4];
        float f[4] = {v.x, v.y, v.z, v.w};
        const int c = c4 * 4;
#pragma unroll
        for (int e = 0; e < 4; ++e) {
            __bf16 h = (__bf16)f[e];
            sXhi[row * CDIM + c + e] = h;
            sXlo[row * CDIM + c + e] = (__bf16)(f[e] - (float)h);
        }
    }
}

__device__ __forceinline__ void update_min(const v8f& acc, float cq, int nIdx,
                                           float bs[8], int bi[8]) {
#pragma unroll
    for (int j = 0; j < 8; ++j) {
        float s = cq + 2.0f * acc[j];   // faithful +2 sign of the reference
        if (s < bs[j] || (s == bs[j] && nIdx < bi[j])) { bs[j] = s; bi[j] = nIdx; }
    }
}

__device__ __forceinline__ void reduce_and_gather(float bs[8], int bi[8],
                                                  const float* __restrict__ cb,
                                                  float* __restrict__ out,
                                                  int mBase, int tid, int wave, int lane,
                                                  float (*sScore)[16], int (*sIdx)[16],
                                                  int* sFinal) {
    // cross-lane argmin within each 16-lane half (C/D layout: lane%16 = N)
#pragma unroll
    for (int j = 0; j < 8; ++j) {
        float s = bs[j];
        int   i = bi[j];
#pragma unroll
        for (int off = 1; off < 16; off <<= 1) {
            float os = __shfl_xor(s, off, 32);
            int   oi = __shfl_xor(i, off, 32);
            if (os < s || (os == s && oi < i)) { s = os; i = oi; }
        }
        bs[j] = s; bi[j] = i;
    }
    if (lane == 0) {
#pragma unroll
        for (int j = 0; j < 8; ++j) { sScore[wave][j] = bs[j]; sIdx[wave][j] = bi[j]; }
    } else if (lane == 16) {
#pragma unroll
        for (int j = 0; j < 8; ++j) { sScore[wave][8 + j] = bs[j]; sIdx[wave][8 + j] = bi[j]; }
    }
    __syncthreads();
    if (tid < 16) {
        float best = 3.0e38f;
        int   bidx = 0;
        for (int w = 0; w < 8; ++w) {
            float s = sScore[w][tid];
            int   i = sIdx[w][tid];
            if (s < best || (s == best && i < bidx)) { best = s; bidx = i; }
        }
        sFinal[tid] = bidx;
    }
    __syncthreads();
    // gather winning codebook rows (exact fp32) to both outputs
    for (int r = 0; r < 16; ++r) {
        const int idx = sFinal[r];
        float4 v = reinterpret_cast<const float4*>(cb + (size_t)idx * CDIM)[tid];
        reinterpret_cast<float4*>(out + (size_t)(mBase + r) * CDIM)[tid] = v;
        reinterpret_cast<float4*>(out + OUT2OFF + (size_t)(mBase + r) * CDIM)[tid] = v;
    }
}

// =====================================================================
// Kernel B (fast path): fused GEMM + argmin + gather, codebook pre-split.
// Each wave: dual N-tiles per K-sweep -> A fragments reused over 6 WMMAs.
// =====================================================================
__global__ __launch_bounds__(256, 1)
void vq_main_pre_kernel(const float* __restrict__ x,
                        const __bf16* __restrict__ cbHi,
                        const __bf16* __restrict__ cbLo,
                        const float* __restrict__ cb,
                        const float* __restrict__ cbsq,
                        float* __restrict__ out) {
    extern __shared__ __align__(16) char smem[];
    __bf16* sXhi = reinterpret_cast<__bf16*>(smem);
    __bf16* sXlo = sXhi + 16 * CDIM;

    __shared__ float sScore[8][16];
    __shared__ int   sIdx[8][16];
    __shared__ int   sFinal[16];

    const int tid   = threadIdx.x;
    const int wave  = tid >> 5;
    const int lane  = tid & 31;
    const int mBase = blockIdx.x * 16;

    stage_x_tile(x, mBase, sXhi, sXlo, tid);
    __syncthreads();

    const int mn  = lane & 15;
    const int sel = lane >> 4;

    float bs[8];
    int   bi[8];
#pragma unroll
    for (int j = 0; j < 8; ++j) { bs[j] = 3.0e38f; bi[j] = 0; }

    for (int tp = wave * 2; tp < KCB / 16; tp += 16) {
        const int n0 = tp * 16;
        const int n1 = n0 + 16;
        const __bf16* bh0 = cbHi + (size_t)(n0 + mn) * CDIM + sel * 8;
        const __bf16* bl0 = cbLo + (size_t)(n0 + mn) * CDIM + sel * 8;
        const __bf16* bh1 = cbHi + (size_t)(n1 + mn) * CDIM + sel * 8;
        const __bf16* bl1 = cbLo + (size_t)(n1 + mn) * CDIM + sel * 8;
        v8f acc0 = {};
        v8f acc1 = {};
        for (int k0 = 0; k0 < CDIM; k0 += 32) {
            v16bf ahi, alo, b0h, b0l, b1h, b1l;
            const __bf16* ph = &sXhi[mn * CDIM + k0 + sel * 8];
            const __bf16* pl = &sXlo[mn * CDIM + k0 + sel * 8];
#pragma unroll
            for (int e = 0; e < 8; ++e) {
                ahi[e]     = ph[e];
                ahi[8 + e] = ph[16 + e];
                alo[e]     = pl[e];
                alo[8 + e] = pl[16 + e];
                b0h[e]     = bh0[k0 + e];
                b0h[8 + e] = bh0[k0 + 16 + e];
                b0l[e]     = bl0[k0 + e];
                b0l[8 + e] = bl0[k0 + 16 + e];
                b1h[e]     = bh1[k0 + e];
                b1h[8 + e] = bh1[k0 + 16 + e];
                b1l[e]     = bl1[k0 + e];
                b1l[8 + e] = bl1[k0 + 16 + e];
            }
            // bf16x3 per tile: hi*hi + hi*lo + lo*hi  (~fp32-accurate dot)
            acc0 = __builtin_amdgcn_wmma_f32_16x16x32_bf16(false, ahi, false, b0h,
                                                           (short)0, acc0, false, false);
            acc0 = __builtin_amdgcn_wmma_f32_16x16x32_bf16(false, ahi, false, b0l,
                                                           (short)0, acc0, false, false);
            acc0 = __builtin_amdgcn_wmma_f32_16x16x32_bf16(false, alo, false, b0h,
                                                           (short)0, acc0, false, false);
            acc1 = __builtin_amdgcn_wmma_f32_16x16x32_bf16(false, ahi, false, b1h,
                                                           (short)0, acc1, false, false);
            acc1 = __builtin_amdgcn_wmma_f32_16x16x32_bf16(false, ahi, false, b1l,
                                                           (short)0, acc1, false, false);
            acc1 = __builtin_amdgcn_wmma_f32_16x16x32_bf16(false, alo, false, b1h,
                                                           (short)0, acc1, false, false);
        }
        update_min(acc0, cbsq[n0 + mn], n0 + mn, bs, bi);
        update_min(acc1, cbsq[n1 + mn], n1 + mn, bs, bi);
    }

    reduce_and_gather(bs, bi, cb, out, mBase, tid, wave, lane, sScore, sIdx, sFinal);
}

// =====================================================================
// Kernel B' (fallback, small workspace): on-the-fly codebook split
// =====================================================================
__global__ __launch_bounds__(256, 1)
void vq_main_fly_kernel(const float* __restrict__ x,
                        const float* __restrict__ cb,
                        const float* __restrict__ cbsq,
                        float* __restrict__ out) {
    extern __shared__ __align__(16) char smem[];
    __bf16* sXhi = reinterpret_cast<__bf16*>(smem);
    __bf16* sXlo = sXhi + 16 * CDIM;

    __shared__ float sScore[8][16];
    __shared__ int   sIdx[8][16];
    __shared__ int   sFinal[16];

    const int tid   = threadIdx.x;
    const int wave  = tid >> 5;
    const int lane  = tid & 31;
    const int mBase = blockIdx.x * 16;

    stage_x_tile(x, mBase, sXhi, sXlo, tid);
    __syncthreads();

    const int mn  = lane & 15;
    const int sel = lane >> 4;

    float bs[8];
    int   bi[8];
#pragma unroll
    for (int j = 0; j < 8; ++j) { bs[j] = 3.0e38f; bi[j] = 0; }

    for (int tile = wave; tile < KCB / 16; tile += 8) {
        const int nBase = tile * 16;
        const float* cbrow = cb + (size_t)(nBase + mn) * CDIM;
        v8f acc = {};
        for (int k0 = 0; k0 < CDIM; k0 += 32) {
            v16bf ahi, alo, bhi, blo;
            const __bf16* ph = &sXhi[mn * CDIM + k0 + sel * 8];
            const __bf16* pl = &sXlo[mn * CDIM + k0 + sel * 8];
#pragma unroll
            for (int e = 0; e < 8; ++e) {
                ahi[e]     = ph[e];
                ahi[8 + e] = ph[16 + e];
                alo[e]     = pl[e];
                alo[8 + e] = pl[16 + e];
            }
            const float* pb = cbrow + k0 + sel * 8;
#pragma unroll
            for (int e = 0; e < 8; ++e) {
                float f0 = pb[e];
                __bf16 h0 = (__bf16)f0;
                bhi[e] = h0;
                blo[e] = (__bf16)(f0 - (float)h0);
                float f1 = pb[16 + e];
                __bf16 h1 = (__bf16)f1;
                bhi[8 + e] = h1;
                blo[8 + e] = (__bf16)(f1 - (float)h1);
            }
            acc = __builtin_amdgcn_wmma_f32_16x16x32_bf16(false, ahi, false, bhi,
                                                          (short)0, acc, false, false);
            acc = __builtin_amdgcn_wmma_f32_16x16x32_bf16(false, ahi, false, blo,
                                                          (short)0, acc, false, false);
            acc = __builtin_amdgcn_wmma_f32_16x16x32_bf16(false, alo, false, bhi,
                                                          (short)0, acc, false, false);
        }
        update_min(acc, cbsq[nBase + mn], nBase + mn, bs, bi);
    }

    reduce_and_gather(bs, bi, cb, out, mBase, tid, wave, lane, sScore, sIdx, sFinal);
}

extern "C" void kernel_launch(void* const* d_in, const int* in_sizes, int n_in,
                              void* d_out, int out_size, void* d_ws, size_t ws_size,
                              hipStream_t stream) {
    const float* z_e_x    = (const float*)d_in[0];   // [16,1024,1024] fp32
    const float* codebook = (const float*)d_in[1];   // [8192,1024]    fp32
    float* out = (float*)d_out;                      // 2 x [16,1024,1024] fp32

    const size_t cbsqBytes  = (size_t)KCB * sizeof(float);                 // 32 KB
    const size_t splitBytes = (size_t)KCB * CDIM * sizeof(__bf16);         // 16 MB each
    const size_t needFast   = cbsqBytes + 2 * splitBytes;                  // ~32 MB
    const size_t ldsBytes   = 2 * 16 * CDIM * sizeof(__bf16);              // 64 KB

    float* cbsq = (float*)d_ws;
    if (ws_size >= needFast) {
        __bf16* cbHi = (__bf16*)((char*)d_ws + cbsqBytes);
        __bf16* cbLo = cbHi + (size_t)KCB * CDIM;
        vq_split_cb_kernel<<<KCB, 256, 0, stream>>>(codebook, cbHi, cbLo, cbsq);
        vq_main_pre_kernel<<<BN_TOT / 16, 256, ldsBytes, stream>>>(
            z_e_x, cbHi, cbLo, codebook, cbsq, out);
    } else {
        vq_cbsq_kernel<<<KCB / 8, 256, 0, stream>>>(codebook, cbsq);
        vq_main_fly_kernel<<<BN_TOT / 16, 256, ldsBytes, stream>>>(
            z_e_x, codebook, cbsq, out);
    }
}
